// Head_56109452755516
// MI455X (gfx1250) — matrix operand order repaired and checked by reading, not compiled, not run on hardware
//
#include <hip/hip_runtime.h>
#include <hip/hip_bf16.h>

typedef __bf16 v16bf __attribute__((ext_vector_type(16)));
typedef float  v8f   __attribute__((ext_vector_type(8)));
typedef unsigned short u16x8 __attribute__((ext_vector_type(8)));

union BF16Frag {
  v16bf bf;
  struct { u16x8 lo; u16x8 hi; } h;
  unsigned short u[16];
  unsigned int d[8];
};

static constexpr int kB  = 16;
static constexpr int kT  = 2048;
static constexpr int kC  = 1024;
static constexpr int kHS = 64;
static constexpr int kQT = kT / 16;          // 128 tiles along T
static constexpr float kScale = 0.125f;      // 1/sqrt(64)

// workspace layout (bytes)
static constexpr size_t OFF_WT = 0;                                   // 393216
static constexpr size_t OFF_QB = OFF_WT + (size_t)3 * kHS * kC * 2;
static constexpr size_t OFF_KB = OFF_QB + (size_t)kB * kT * kHS * 2;
static constexpr size_t OFF_VT = OFF_KB + (size_t)kB * kT * kHS * 2;
static constexpr size_t OFF_M  = OFF_VT + (size_t)kB * kT * kHS * 2;
static constexpr size_t OFF_L  = OFF_M + (size_t)kB * kT * 4;

__device__ __forceinline__ unsigned short f2bf(float f) {
  unsigned int u = __float_as_uint(f) + 0x8000u;  // round-half-up
  return (unsigned short)(u >> 16);
}

// two fp32 -> packed bf16x2 dword: 2x v_add + 1x v_perm_b32
__device__ __forceinline__ unsigned int pack2bf(float lo, float hi) {
  unsigned int ul = __float_as_uint(lo) + 0x8000u;
  unsigned int uh = __float_as_uint(hi) + 0x8000u;
  return __builtin_amdgcn_perm(uh, ul, 0x07060302u);
}

__device__ __forceinline__ v8f zero8() {
  v8f z;
#pragma unroll
  for (int i = 0; i < 8; ++i) z[i] = 0.0f;
  return z;
}

// A-fragment (16x32 bf16): lane = row (lane&15); elems 0..7 at off0, 8..15 at off1.
__device__ __forceinline__ v16bf loadA_bf16(const unsigned short* row, int off0, int off1) {
  BF16Frag f;
  f.h.lo = *reinterpret_cast<const u16x8*>(row + off0);
  f.h.hi = *reinterpret_cast<const u16x8*>(row + off1);
  return f.bf;
}

// B-fragment (32x16 bf16): lane = col (lane&15); 16 contiguous K values at off.
__device__ __forceinline__ v16bf loadB_bf16(const unsigned short* base, int off) {
  BF16Frag f;
  f.h.lo = *reinterpret_cast<const u16x8*>(base + off);
  f.h.hi = *reinterpret_cast<const u16x8*>(base + off + 8);
  return f.bf;
}

// ---------------------------------------------------------------------------
// 1) Convert + transpose weights: WT[w][h][c] = bf16(W_w[c][h])
// ---------------------------------------------------------------------------
__global__ __launch_bounds__(256) void prep_weights_kernel(
    const float* __restrict__ Wq, const float* __restrict__ Wk,
    const float* __restrict__ Wv, unsigned short* __restrict__ WT) {
  int idx = blockIdx.x * blockDim.x + threadIdx.x;
  if (idx >= 3 * kHS * kC) return;
  int w = idx / (kHS * kC);
  int r = idx - w * (kHS * kC);
  int h = r / kC;
  int c = r - h * kC;
  const float* W = (w == 0) ? Wq : (w == 1) ? Wk : Wv;
  WT[idx] = f2bf(W[c * kHS + h]);
}

// ---------------------------------------------------------------------------
// 2) Projections: q,k -> [B*T, HS] bf16 row-major ; v -> vT [B][HS][T] bf16
// ---------------------------------------------------------------------------
__global__ __launch_bounds__(128) void proj_kernel(
    const float* __restrict__ x, const unsigned short* __restrict__ WT,
    unsigned short* __restrict__ qb, unsigned short* __restrict__ kb,
    unsigned short* __restrict__ vT) {
  const int lane   = threadIdx.x & 31;
  const int wave   = __builtin_amdgcn_readfirstlane((blockIdx.x << 2) + (threadIdx.x >> 5));
  const int outSel = wave % 3;            // scalar
  const int tile   = wave / 3;            // scalar, 0..2047
  const int col16  = lane & 15;
  const int half   = lane >> 4;

  const float* xrow = x + (size_t)(tile * 16 + col16) * kC;
  const unsigned short* Wbase = WT + (size_t)outSel * kHS * kC;

  v8f acc[4];
#pragma unroll
  for (int n = 0; n < 4; ++n) acc[n] = zero8();

  for (int cc = 0; cc < kC; cc += 32) {
    // B fragments first (8 clustered b128 loads)
    v16bf b[4];
#pragma unroll
    for (int n = 0; n < 4; ++n) {
      const unsigned short* wrow = Wbase + (size_t)(n * 16 + col16) * kC;
      b[n] = loadB_bf16(wrow, cc + half * 16);
    }
    // A fragment: fp32 x converted to bf16 with packed v_perm
    BF16Frag a;
    const float4* p0 = reinterpret_cast<const float4*>(xrow + cc + half * 16);
    const float4* p1 = reinterpret_cast<const float4*>(xrow + cc + half * 16 + 8);
    float4 f0 = p0[0], f1 = p0[1], f2 = p1[0], f3 = p1[1];
    a.d[0] = pack2bf(f0.x, f0.y); a.d[1] = pack2bf(f0.z, f0.w);
    a.d[2] = pack2bf(f1.x, f1.y); a.d[3] = pack2bf(f1.z, f1.w);
    a.d[4] = pack2bf(f2.x, f2.y); a.d[5] = pack2bf(f2.z, f2.w);
    a.d[6] = pack2bf(f3.x, f3.y); a.d[7] = pack2bf(f3.z, f3.w);

#pragma unroll
    for (int n = 0; n < 4; ++n)
      acc[n] = __builtin_amdgcn_wmma_f32_16x16x32_bf16(
          false, a.bf, false, b[n], (short)0, acc[n], false, false);
  }

  // store: single scalar branch on outSel
  if (outSel == 0) {
#pragma unroll
    for (int n = 0; n < 4; ++n)
#pragma unroll
      for (int r = 0; r < 8; ++r)
        qb[(size_t)(tile * 16 + r + half * 8) * kHS + n * 16 + col16] = f2bf(acc[n][r]);
  } else if (outSel == 1) {
#pragma unroll
    for (int n = 0; n < 4; ++n)
#pragma unroll
      for (int r = 0; r < 8; ++r)
        kb[(size_t)(tile * 16 + r + half * 8) * kHS + n * 16 + col16] = f2bf(acc[n][r]);
  } else {
#pragma unroll
    for (int n = 0; n < 4; ++n)
#pragma unroll
      for (int r = 0; r < 8; ++r) {
        int rowD = tile * 16 + r + half * 8;
        int bb = rowD >> 11;
        int tt = rowD & (kT - 1);
        vT[((size_t)bb * kHS + n * 16 + col16) * kT + tt] = f2bf(acc[n][r]);
      }
  }
}

// ---------------------------------------------------------------------------
// 3) Column-softmax stats (softmax over QUERY axis): one wave per (b, k-tile).
// ---------------------------------------------------------------------------
__global__ __launch_bounds__(128) void stats_kernel(
    const unsigned short* __restrict__ qb, const unsigned short* __restrict__ kb,
    float* __restrict__ mArr, float* __restrict__ lrecArr) {
  const int lane  = threadIdx.x & 31;
  const int wave  = __builtin_amdgcn_readfirstlane((blockIdx.x << 2) + (threadIdx.x >> 5));
  const int b     = wave >> 7;     // scalar
  const int ktile = wave & 127;    // scalar
  const int col16 = lane & 15;
  const int half  = lane >> 4;

  const unsigned short* krow = kb + (size_t)(b * kT + ktile * 16 + col16) * kHS;
  v16bf bk0 = loadB_bf16(krow, 0 + half * 16);
  v16bf bk1 = loadB_bf16(krow, 32 + half * 16);

  float mrun = -3.0e38f;
  float lrun = 0.0f;

  const unsigned short* qbB = qb + (size_t)b * kT * kHS;
  for (int qt = ktile; qt < kQT; ++qt) {
    const unsigned short* qrow = qbB + (size_t)(qt * 16 + col16) * kHS;
    v16bf a0 = loadA_bf16(qrow, half * 16, half * 16 + 8);
    v16bf a1 = loadA_bf16(qrow, 32 + half * 16, 40 + half * 16);
    v8f s = zero8();
    s = __builtin_amdgcn_wmma_f32_16x16x32_bf16(false, a0, false, bk0, (short)0, s, false, false);
    s = __builtin_amdgcn_wmma_f32_16x16x32_bf16(false, a1, false, bk1, (short)0, s, false, false);

    float sv[8];
#pragma unroll
    for (int r = 0; r < 8; ++r) {
      float v = s[r] * kScale;
      if (qt == ktile && (r + half * 8) < col16) v = -3.0e38f;   // causal mask
      sv[r] = v;
    }
    float tmax = sv[0];
#pragma unroll
    for (int r = 1; r < 8; ++r) tmax = fmaxf(tmax, sv[r]);
    tmax = fmaxf(tmax, __shfl_xor(tmax, 16, 32));
    float newm = fmaxf(mrun, tmax);
    float p = 0.0f;
#pragma unroll
    for (int r = 0; r < 8; ++r) p += __expf(sv[r] - newm);
    p += __shfl_xor(p, 16, 32);
    lrun = lrun * __expf(mrun - newm) + p;
    mrun = newm;
  }
  if (half == 0) {
    int kg = b * kT + ktile * 16 + col16;
    mArr[kg]    = mrun;
    lrecArr[kg] = 1.0f / lrun;
  }
}

// ---------------------------------------------------------------------------
// 4) Output: one wave per (b, q-tile); recompute S in 32-wide k chunks.
// ---------------------------------------------------------------------------
__global__ __launch_bounds__(128) void attn_out_kernel(
    const unsigned short* __restrict__ qb, const unsigned short* __restrict__ kb,
    const unsigned short* __restrict__ vT, const float* __restrict__ mArr,
    const float* __restrict__ lrecArr, float* __restrict__ out) {
  __shared__ unsigned short Pbuf[4][16 * 32];
  const int lane  = threadIdx.x & 31;
  const int wslot = __builtin_amdgcn_readfirstlane(threadIdx.x >> 5);
  const int wave  = __builtin_amdgcn_readfirstlane((blockIdx.x << 2) + (threadIdx.x >> 5));
  const int b     = wave >> 7;     // scalar
  const int qtile = wave & 127;    // scalar
  const int col16 = lane & 15;
  const int half  = lane >> 4;
  unsigned short* pb = &Pbuf[wslot][0];

  const unsigned short* qrow = qb + (size_t)(b * kT + qtile * 16 + col16) * kHS;
  v16bf aq0 = loadA_bf16(qrow, half * 16, half * 16 + 8);
  v16bf aq1 = loadA_bf16(qrow, 32 + half * 16, 40 + half * 16);

  v8f o[4];
#pragma unroll
  for (int n = 0; n < 4; ++n) o[n] = zero8();

  const unsigned short* kbB = kb + (size_t)b * kT * kHS;
  const unsigned short* vTB = vT + (size_t)b * kHS * kT;
  const float* mB = mArr + (size_t)b * kT;
  const float* lB = lrecArr + (size_t)b * kT;

  const int nchunk = (qtile >> 1) + 1;
  for (int ch = 0; ch < nchunk; ++ch) {
    const int k0 = ch * 32;
    const int kg0 = k0 + col16;
    const int kg1 = k0 + 16 + col16;

    // K fragments for both sub-tiles (clustered loads)
    const unsigned short* krow0 = kbB + (size_t)kg0 * kHS;
    const unsigned short* krow1 = kbB + (size_t)kg1 * kHS;
    v16bf bk00 = loadB_bf16(krow0, half * 16);
    v16bf bk01 = loadB_bf16(krow0, 32 + half * 16);
    v16bf bk10 = loadB_bf16(krow1, half * 16);
    v16bf bk11 = loadB_bf16(krow1, 32 + half * 16);

    // V fragments + softmax stats issued early: latency hidden under WMMA/exp
    v16bf bv[4];
#pragma unroll
    for (int n = 0; n < 4; ++n)
      bv[n] = loadB_bf16(vTB + (size_t)(n * 16 + col16) * kT, k0 + half * 16);
    float mcol0 = mB[kg0], lcol0 = lB[kg0];
    float mcol1 = mB[kg1], lcol1 = lB[kg1];

    v8f s0 = zero8(), s1 = zero8();
    s0 = __builtin_amdgcn_wmma_f32_16x16x32_bf16(false, aq0, false, bk00, (short)0, s0, false, false);
    s0 = __builtin_amdgcn_wmma_f32_16x16x32_bf16(false, aq1, false, bk01, (short)0, s0, false, false);
    s1 = __builtin_amdgcn_wmma_f32_16x16x32_bf16(false, aq0, false, bk10, (short)0, s1, false, false);
    s1 = __builtin_amdgcn_wmma_f32_16x16x32_bf16(false, aq1, false, bk11, (short)0, s1, false, false);

#pragma unroll
    for (int r = 0; r < 8; ++r) {
      int qg = qtile * 16 + r + half * 8;
      float w0 = 0.0f, w1 = 0.0f;
      if (qg >= kg0) w0 = __expf(s0[r] * kScale - mcol0) * lcol0;
      if (qg >= kg1) w1 = __expf(s1[r] * kScale - mcol1) * lcol1;
      pb[(r + half * 8) * 32 + col16]      = f2bf(w0);
      pb[(r + half * 8) * 32 + 16 + col16] = f2bf(w1);
    }
    // DS ops are in-order per wave; wait for stores before cross-lane reload
    asm volatile("s_wait_dscnt 0x0" ::: "memory");

    const unsigned short* prow = pb + col16 * 32;
    BF16Frag pa;
    pa.h.lo = *reinterpret_cast<const u16x8*>(prow + half * 16);
    pa.h.hi = *reinterpret_cast<const u16x8*>(prow + half * 16 + 8);
#pragma unroll
    for (int n = 0; n < 4; ++n)
      o[n] = __builtin_amdgcn_wmma_f32_16x16x32_bf16(
          false, pa.bf, false, bv[n], (short)0, o[n], false, false);
  }

#pragma unroll
  for (int n = 0; n < 4; ++n)
#pragma unroll
    for (int r = 0; r < 8; ++r) {
      int rowD = b * kT + qtile * 16 + r + half * 8;
      out[(size_t)rowD * kHS + n * 16 + col16] = o[n][r];
    }
}

// ---------------------------------------------------------------------------
extern "C" void kernel_launch(void* const* d_in, const int* in_sizes, int n_in,
                              void* d_out, int out_size, void* d_ws, size_t ws_size,
                              hipStream_t stream) {
  const float* x  = (const float*)d_in[0];
  const float* Wq = (const float*)d_in[1];
  const float* Wk = (const float*)d_in[2];
  const float* Wv = (const float*)d_in[3];
  float* out = (float*)d_out;

  char* ws = (char*)d_ws;
  unsigned short* WT   = (unsigned short*)(ws + OFF_WT);
  unsigned short* qb   = (unsigned short*)(ws + OFF_QB);
  unsigned short* kb   = (unsigned short*)(ws + OFF_KB);
  unsigned short* vT   = (unsigned short*)(ws + OFF_VT);
  float*          mArr = (float*)(ws + OFF_M);
  float*          lrec = (float*)(ws + OFF_L);

  prep_weights_kernel<<<(3 * kHS * kC) / 256, 256, 0, stream>>>(Wq, Wk, Wv, WT);
  proj_kernel<<<1536, 128, 0, stream>>>(x, WT, qb, kb, vT);
  stats_kernel<<<512, 128, 0, stream>>>(qb, kb, mArr, lrec);
  attn_out_kernel<<<512, 128, 0, stream>>>(qb, kb, vT, mArr, lrec, out);
}
